// GATModel_68178310856974
// MI455X (gfx1250) — matrix operand order repaired and checked by reading, not compiled, not run on hardware
//
#include <hip/hip_runtime.h>
#include <math.h>

#define NEG_SLOPE 0.2f

typedef __attribute__((ext_vector_type(16))) _Float16 v16h;
typedef __attribute__((ext_vector_type(8)))  float    v8f;
typedef __attribute__((ext_vector_type(4)))  float    v4f;

// ---------------------------------------------------------------------------
// float atomic max via signed-int / unsigned-int ordering trick (native atomics)
// ---------------------------------------------------------------------------
__device__ __forceinline__ void atomicMaxF32(float* addr, float val) {
  if (val >= 0.0f) atomicMax((int*)addr, __float_as_int(val));
  else             atomicMin((unsigned int*)addr, __float_as_uint(val));
}

// ---------------------------------------------------------------------------
// Pack W[din, dout] (f32, row-major) into f16 "B-fragment order":
//   Wp[((kt*colTiles + nt)*32 + lane)*16 + j] = W[kt*32 + (lane>>4)*8 + (j&7) + (j>>3)*16][nt*16 + (lane&15)]
// so each GEMM lane reads its 16 halves as one aligned 32-byte load.
// ---------------------------------------------------------------------------
__global__ void pack_w_f16(const float* __restrict__ W, _Float16* __restrict__ Wp,
                           int din, int dout) {
  const int total = din * dout;
  const int idx = blockIdx.x * blockDim.x + threadIdx.x;
  if (idx >= total) return;
  const int colTiles = dout >> 4;
  const int j    = idx & 15;
  const int lane = (idx >> 4) & 31;
  const int rest = idx >> 9;
  const int nt   = rest % colTiles;
  const int kt   = rest / colTiles;
  const int k    = kt * 32 + (lane >> 4) * 8 + (j & 7) + (j >> 3) * 16;
  const int col  = nt * 16 + (lane & 15);
  Wp[idx] = (_Float16)W[(size_t)k * dout + col];
}

// ---------------------------------------------------------------------------
// GEMM: Y[n, dout] = X[n, din] @ W[din, dout], one wave per 16x16 tile,
// K-loop in steps of 32 using v_wmma_f32_16x16x32_f16 (f32 accumulation).
// A loaded as 4x b128 f32 + cvt; B loaded pre-packed f16 (2x b128).
// ---------------------------------------------------------------------------
__global__ void gemm_wmma_f16(const float* __restrict__ X,
                              const _Float16* __restrict__ Wp,
                              float* __restrict__ Y,
                              int n, int din, int dout) {
  const int lane  = threadIdx.x & 31;
  const int wave  = (blockIdx.x * blockDim.x + threadIdx.x) >> 5;
  const int colTiles = dout >> 4;
  const int numTiles = (n >> 4) * colTiles;
  if (wave >= numTiles) return;                 // wave-uniform: EXEC stays full
  const int mt = wave / colTiles;
  const int nt = wave - mt * colTiles;
  const int row0 = mt << 4, col0 = nt << 4;

  const int m    = lane & 15;                   // A row / B,C column
  const int half = lane >> 4;                   // lane half selects K sub-block
  const int kbA  = half * 8;

  v8f acc = {};
  const int kSteps = din >> 5;
  for (int kt = 0; kt < kSteps; ++kt) {
    // --- A fragment: row m; two contiguous 8-float runs -> 4x b128 loads ---
    const float* xr = X + (size_t)(row0 + m) * din + (kt << 5);
    const v4f* xv = (const v4f*)(xr + kbA);     // 32B-aligned
    const v4f p0 = xv[0], p1 = xv[1];           // K = kbA .. kbA+7
    const v4f q0 = xv[4], q1 = xv[5];           // K = 16+kbA .. 16+kbA+7
    v16h a;
    a[0] = (_Float16)p0[0]; a[1] = (_Float16)p0[1];
    a[2] = (_Float16)p0[2]; a[3] = (_Float16)p0[3];
    a[4] = (_Float16)p1[0]; a[5] = (_Float16)p1[1];
    a[6] = (_Float16)p1[2]; a[7] = (_Float16)p1[3];
    a[8]  = (_Float16)q0[0]; a[9]  = (_Float16)q0[1];
    a[10] = (_Float16)q0[2]; a[11] = (_Float16)q0[3];
    a[12] = (_Float16)q1[0]; a[13] = (_Float16)q1[1];
    a[14] = (_Float16)q1[2]; a[15] = (_Float16)q1[3];
    // --- B fragment: pre-packed, one aligned 32-byte read per lane ---
    const v16h b = *(const v16h*)(Wp + (((size_t)kt * colTiles + nt) * 32 + lane) * 16);
    acc = __builtin_amdgcn_wmma_f32_16x16x32_f16(false, a, false, b,
                                                 (short)0, acc, false, false);
  }
  // --- C/D: lane = column m, VGPR i = row half*8 + i ---
#pragma unroll
  for (int i = 0; i < 8; ++i)
    Y[(size_t)(row0 + half * 8 + i) * dout + col0 + m] = acc[i];
}

// ---------------------------------------------------------------------------
// el/er: per-node, per-head dot products of feat with al/ar.
// Block = heads*d threads (128 or 64), one node per block, LDS tree-reduce
// inside each head's d-sized (power-of-two) segment.
// ---------------------------------------------------------------------------
__global__ void attn_scores(const float* __restrict__ feat,
                            const float* __restrict__ al,
                            const float* __restrict__ ar,
                            float* __restrict__ el, float* __restrict__ er,
                            int heads, int d) {
  const int node = blockIdx.x;
  const int tid  = threadIdx.x;
  const int hd   = heads * d;
  __shared__ float sl[128];
  __shared__ float sr[128];
  const float f = feat[(size_t)node * hd + tid];
  sl[tid] = f * al[tid];
  sr[tid] = f * ar[tid];
  __syncthreads();
  for (int off = d >> 1; off > 0; off >>= 1) {
    if ((tid & (d - 1)) < off) {
      sl[tid] += sl[tid + off];
      sr[tid] += sr[tid + off];
    }
    __syncthreads();
  }
  if ((tid & (d - 1)) == 0) {
    const int h = tid / d;
    el[(size_t)node * heads + h] = sl[tid];
    er[(size_t)node * heads + h] = sr[tid];
  }
}

// ---------------------------------------------------------------------------
__global__ void init_scores(float* __restrict__ m, float* __restrict__ s, int n) {
  const int i = blockIdx.x * blockDim.x + threadIdx.x;
  if (i < n) { m[i] = -INFINITY; s[i] = 0.0f; }
}

// out[i] = bias[i % hd], vectorized by 4 (hd multiple of 4).
__global__ void init_out4(float* __restrict__ out, const float* __restrict__ bias,
                          int nvec, int hdvec) {
  const int i = blockIdx.x * blockDim.x + threadIdx.x;
  if (i >= nvec) return;
  ((v4f*)out)[i] = ((const v4f*)bias)[i % hdvec];
}

__device__ __forceinline__ float lrelu(float v) {
  return v > 0.0f ? v : v * NEG_SLOPE;
}

// Pass 1: per-(edge,head) segment max over destinations.
__global__ void edge_max(const int* __restrict__ src, const int* __restrict__ dst,
                         const float* __restrict__ el, const float* __restrict__ er,
                         float* __restrict__ mx, int e, int heads) {
  const int i = blockIdx.x * blockDim.x + threadIdx.x;
  if (i >= e * heads) return;
  const int ed = i / heads, h = i - ed * heads;
  const int s = src[ed], dn = dst[ed];
  const float v = lrelu(el[(size_t)s * heads + h] + er[(size_t)dn * heads + h]);
  atomicMaxF32(&mx[(size_t)dn * heads + h], v);
}

// Pass 2: per-(edge,head) exp-sum.
__global__ void edge_expsum(const int* __restrict__ src, const int* __restrict__ dst,
                            const float* __restrict__ el, const float* __restrict__ er,
                            const float* __restrict__ mx, float* __restrict__ sm,
                            int e, int heads) {
  const int i = blockIdx.x * blockDim.x + threadIdx.x;
  if (i >= e * heads) return;
  const int ed = i / heads, h = i - ed * heads;
  const int s = src[ed], dn = dst[ed];
  const float v = lrelu(el[(size_t)s * heads + h] + er[(size_t)dn * heads + h]);
  atomicAdd(&sm[(size_t)dn * heads + h], expf(v - mx[(size_t)dn * heads + h]));
}

// Pass 3: per-(edge, 4 feature elements): one b128 gather of feat[src], alpha
// computed once per thread, 4 native f32 atomic adds into out[dst].
__global__ void edge_scatter4(const int* __restrict__ src, const int* __restrict__ dst,
                              const float* __restrict__ el, const float* __restrict__ er,
                              const float* __restrict__ mx, const float* __restrict__ sm,
                              const float* __restrict__ feat, float* __restrict__ out,
                              int e, int heads, int d) {
  const int hd  = heads * d;
  const int vec = hd >> 2;                         // 4-float groups per edge
  const int i   = blockIdx.x * blockDim.x + threadIdx.x;
  if (i >= e * vec) return;
  const int ed = i / vec;
  const int t4 = (i - ed * vec) << 2;              // element offset (mult of 4)
  const int h  = t4 / d;                           // d is 32 or 64: group stays in one head
  const int s  = src[ed], dn = dst[ed];
  const float v     = lrelu(el[(size_t)s * heads + h] + er[(size_t)dn * heads + h]);
  const float alpha = expf(v - mx[(size_t)dn * heads + h]) / sm[(size_t)dn * heads + h];
  const v4f f = *(const v4f*)(feat + (size_t)s * hd + t4);
  float* o = out + (size_t)dn * hd + t4;
  atomicAdd(o + 0, alpha * f[0]);
  atomicAdd(o + 1, alpha * f[1]);
  atomicAdd(o + 2, alpha * f[2]);
  atomicAdd(o + 3, alpha * f[3]);
}

// ELU in place, vectorized by 4.
__global__ void elu4_kernel(float* __restrict__ x, int nvec) {
  const int i = blockIdx.x * blockDim.x + threadIdx.x;
  if (i >= nvec) return;
  v4f v = ((v4f*)x)[i];
#pragma unroll
  for (int c = 0; c < 4; ++c) v[c] = v[c] > 0.0f ? v[c] : (expf(v[c]) - 1.0f);
  ((v4f*)x)[i] = v;
}

// ---------------------------------------------------------------------------
extern "C" void kernel_launch(void* const* d_in, const int* in_sizes, int n_in,
                              void* d_out, int out_size, void* d_ws, size_t ws_size,
                              hipStream_t stream) {
  const float* h   = (const float*)d_in[0];
  const int*   src = (const int*)  d_in[1];
  const int*   dst = (const int*)  d_in[2];
  const float* W1  = (const float*)d_in[3];
  const float* al1 = (const float*)d_in[4];
  const float* ar1 = (const float*)d_in[5];
  const float* b1  = (const float*)d_in[6];
  const float* W2  = (const float*)d_in[7];
  const float* al2 = (const float*)d_in[8];
  const float* ar2 = (const float*)d_in[9];
  const float* b2  = (const float*)d_in[10];
  const float* W3  = (const float*)d_in[11];
  const float* al3 = (const float*)d_in[12];
  const float* ar3 = (const float*)d_in[13];
  const float* b3  = (const float*)d_in[14];

  const int N = 50000, E = 800000;

  // Workspace (floats): bufX[N*128] | bufF[N*128] | el,er,mx,sm [N*4 each] | Wp (f16)
  float* bufX = (float*)d_ws;
  float* bufF = bufX + (size_t)N * 128;
  float* el   = bufF + (size_t)N * 128;
  float* er   = el + (size_t)N * 4;
  float* mx   = er + (size_t)N * 4;
  float* sm   = mx + (size_t)N * 4;
  _Float16* Wp = (_Float16*)(sm + (size_t)N * 4);   // <= 128*128 halves (32 KB)

  auto run_layer = [&](const float* x, const float* W, const float* al,
                       const float* ar, const float* b, float* feat, float* out,
                       int din, int heads, int d) {
    const int dout  = heads * d;
    const int waves = (N / 16) * (dout / 16);
    pack_w_f16<<<(din * dout + 255) / 256, 256, 0, stream>>>(W, Wp, din, dout);
    gemm_wmma_f16<<<(waves + 7) / 8, 256, 0, stream>>>(x, Wp, feat, N, din, dout);
    attn_scores<<<N, dout, 0, stream>>>(feat, al, ar, el, er, heads, d);
    init_scores<<<(N * heads + 255) / 256, 256, 0, stream>>>(mx, sm, N * heads);
    init_out4<<<(N * dout / 4 + 255) / 256, 256, 0, stream>>>(out, b, N * dout / 4, dout / 4);
    edge_max<<<(E * heads + 255) / 256, 256, 0, stream>>>(src, dst, el, er, mx, E, heads);
    edge_expsum<<<(E * heads + 255) / 256, 256, 0, stream>>>(src, dst, el, er, mx, sm, E, heads);
    const int tot = E * (dout / 4);   // <= 25.6M
    edge_scatter4<<<(tot + 255) / 256, 256, 0, stream>>>(src, dst, el, er, mx, sm,
                                                         feat, out, E, heads, d);
  };

  // Layer 1: h -> bufF (feat) -> bufX (out), then ELU in place.
  run_layer(h, W1, al1, ar1, b1, bufF, bufX, 128, 4, 32);
  elu4_kernel<<<(N * 32 + 255) / 256, 256, 0, stream>>>(bufX, N * 32);

  // Layer 2: bufX is dead after its GEMM, so it doubles as the accumulator.
  run_layer(bufX, W2, al2, ar2, b2, bufF, bufX, 128, 4, 32);
  elu4_kernel<<<(N * 32 + 255) / 256, 256, 0, stream>>>(bufX, N * 32);

  // Layer 3: heads=1, d=64; mean over 1 head is identity -> write d_out directly.
  run_layer(bufX, W3, al3, ar3, b3, bufF, (float*)d_out, 128, 1, 64);
}
